// MultiHeadedAttention_55250459295870
// MI455X (gfx1250) — matrix operand order repaired and checked by reading, not compiled, not run on hardware
//
#include <hip/hip_runtime.h>

// Problem constants (from reference): B=2, S=2048, D=1024, H=16, HD=64
#define S_LEN 2048
#define D_DIM 1024
#define NH    16
#define HD_   64
#define BB    2

typedef unsigned short u16;
typedef __attribute__((ext_vector_type(16))) _Float16 v16h;
typedef __attribute__((ext_vector_type(8)))  float    v8f;

// 16x32 A-frag / 32x16 B-frag: 16 f16 per lane = 32 bytes = two uint4 chunks
union Frag  { uint4 q[2]; v16h v; };
union Pack8 { u16 h[8];  uint4 q; };

static __device__ __forceinline__ u16 f2h(float f) {
  union { _Float16 h; u16 u; } cv;
  cv.h = (_Float16)f;                 // v_cvt_f16_f32 (RNE)
  return cv.u;
}

static __device__ __forceinline__ v8f wmma_f16(Frag a, Frag b, v8f c) {
  // D = A(16x32 f16) * B(32x16 f16) + C(16x16 f32)
  return __builtin_amdgcn_wmma_f32_16x16x32_f16(
      /*neg_a=*/false, a.v, /*neg_b=*/false, b.v,
      /*c_mod=*/(short)0, c, /*reuse_a=*/false, /*reuse_b=*/false);
}

// Async DMA: 16B global -> LDS, tracked by ASYNCcnt (CDNA5 path).
// Generic pointers to __shared__ carry the LDS byte offset in their low 32 bits.
static __device__ __forceinline__ void async_b128(unsigned lds_off,
                                                  unsigned long long gaddr) {
  asm volatile("global_load_async_to_lds_b128 %0, %1, off"
               :: "v"(lds_off), "v"(gaddr) : "memory");
}

// ---------------------------------------------------------------- kernel 0a
// fp32 -> fp16 bulk convert (x), 4 elements/thread
__global__ void cvt_f32_f16(const float* __restrict__ src, u16* __restrict__ dst, int n) {
  int i = (blockIdx.x * blockDim.x + threadIdx.x) * 4;
  if (i + 3 < n) {
    float4 f = *(const float4*)(src + i);
    uint2 r;
    r.x = (unsigned)f2h(f.x) | ((unsigned)f2h(f.y) << 16);
    r.y = (unsigned)f2h(f.z) | ((unsigned)f2h(f.w) << 16);
    *(uint2*)(dst + i) = r;
  }
}

// ---------------------------------------------------------------- kernel 0b
// W [in(k) x out(n)] fp32 -> Wt [n x k] fp16  (LDS-tiled transpose)
__global__ void wtrans(const float* __restrict__ Wq, const float* __restrict__ Wk,
                       const float* __restrict__ Wv, u16* __restrict__ Wt) {
  const float* W = (blockIdx.z == 0) ? Wq : (blockIdx.z == 1) ? Wk : Wv;
  u16* dst = Wt + (size_t)blockIdx.z * D_DIM * D_DIM;
  __shared__ float tile[32][33];
  int x = blockIdx.x * 32 + threadIdx.x;   // n (cols of W, coalesced read)
  int y0 = blockIdx.y * 32;                // k
  for (int j = threadIdx.y; j < 32; j += 8)
    tile[j][threadIdx.x] = W[(size_t)(y0 + j) * D_DIM + x];
  __syncthreads();
  int xo = blockIdx.y * 32 + threadIdx.x;  // k (coalesced write)
  int yo0 = blockIdx.x * 32;               // n
  for (int j = threadIdx.y; j < 32; j += 8)
    dst[(size_t)(yo0 + j) * D_DIM + xo] = f2h(tile[threadIdx.x][j]);
}

// ---------------------------------------------------------------- kernel 1
// Fused QKV projection: [4096 x 1024] x Wt^T + bias.
// z=0 -> Q f16 [B*S, D] row-major; z=1 -> K same; z=2 -> V transposed [B, D, S].
// Block tile 128(M) x 64(N), 8 waves (4 M-bands x 2 N-bands), wave tile 32x32.
__global__ void __launch_bounds__(256)
qkv_gemm(const u16* __restrict__ xh, const u16* __restrict__ Wt,
         const float* __restrict__ bq, const float* __restrict__ bk,
         const float* __restrict__ bv,
         u16* __restrict__ Qh, u16* __restrict__ Kh, u16* __restrict__ Vt) {
  const int z = blockIdx.z;
  const u16* W = Wt + (size_t)z * D_DIM * D_DIM;
  const float* bias = (z == 0) ? bq : (z == 1) ? bk : bv;

  const int lane = threadIdx.x & 31;
  const int wave = threadIdx.x >> 5;
  const int g = lane >> 4, l15 = lane & 15;
  const int m0 = blockIdx.x * 128 + (wave & 3) * 32;
  const int n0 = blockIdx.y * 64 + (wave >> 2) * 32;

  v8f acc[2][2];
#pragma unroll
  for (int mi = 0; mi < 2; ++mi)
#pragma unroll
    for (int ni = 0; ni < 2; ++ni)
#pragma unroll
      for (int r = 0; r < 8; ++r) acc[mi][ni][r] = 0.f;

  for (int k0 = 0; k0 < D_DIM; k0 += 32) {
    Frag a[2], bf[2];
#pragma unroll
    for (int mi = 0; mi < 2; ++mi) {
      const u16* p = xh + (size_t)(m0 + mi * 16 + l15) * D_DIM + k0 + 8 * g;
      a[mi].q[0] = *(const uint4*)p;          // K = k0+8g .. +7
      a[mi].q[1] = *(const uint4*)(p + 16);   // K = k0+16+8g .. +7
    }
#pragma unroll
    for (int ni = 0; ni < 2; ++ni) {
      const u16* p = W + (size_t)(n0 + ni * 16 + l15) * D_DIM + k0 + 8 * g;
      bf[ni].q[0] = *(const uint4*)p;         // K = k0+8g .. +7
      bf[ni].q[1] = *(const uint4*)(p + 16);  // K = k0+16+8g .. +7
    }
#pragma unroll
    for (int mi = 0; mi < 2; ++mi)
#pragma unroll
      for (int ni = 0; ni < 2; ++ni)
        acc[mi][ni] = wmma_f16(a[mi], bf[ni], acc[mi][ni]);
  }

#pragma unroll
  for (int mi = 0; mi < 2; ++mi) {
#pragma unroll
    for (int ni = 0; ni < 2; ++ni) {
      const int n = n0 + ni * 16 + l15;
      const float bn = bias[n];
      if (z < 2) {
        u16* dst = (z == 0) ? Qh : Kh;
#pragma unroll
        for (int r = 0; r < 8; ++r) {
          const int mg = m0 + mi * 16 + 8 * g + r;       // C-frag row = r + 8g
          dst[(size_t)mg * D_DIM + n] = f2h(acc[mi][ni][r] + bn);
        }
      } else {
        // V transposed: 8 consecutive rows per lane -> one 16B store
        Pack8 pk;
#pragma unroll
        for (int r = 0; r < 8; ++r) pk.h[r] = f2h(acc[mi][ni][r] + bn);
        const int mg = m0 + mi * 16 + 8 * g;
        const int bidx = mg >> 11;             // / S_LEN
        const int s = mg & (S_LEN - 1);
        *(uint4*)(Vt + ((size_t)(bidx * D_DIM + n)) * S_LEN + s) = pk.q;
      }
    }
  }
}

// ---------------------------------------------------------------- kernel 2
// Flash attention with online softmax. Reference quirk preserved:
// masked scores are set to 0 (NOT -inf) before softmax.
// Mask tiles are DMA'd global->LDS with global_load_async_to_lds_b128
// (double-buffered per wave, synced with s_wait_asynccnt).
// Grid: (S/64, B*H). 4 waves/block, each wave owns 16 query rows.
__global__ void __launch_bounds__(128)
attn(const u16* __restrict__ Qh, const u16* __restrict__ Kh,
     const u16* __restrict__ Vt, const unsigned char* __restrict__ mask,
     float* __restrict__ out) {
  __shared__ u16 shP[4 * 16 * 64];                 // per-wave 16x64 f16 P tile
  __shared__ unsigned char shM[4 * 2 * 16 * 64];   // per-wave double-buffered mask tile
  const int lane = threadIdx.x & 31;
  const int wave = threadIdx.x >> 5;
  const int g = lane >> 4, l15 = lane & 15;
  const int bh = blockIdx.y;
  const int b = bh >> 4, h = bh & 15;
  const int s0 = blockIdx.x * 64 + wave * 16;      // query rows [s0, s0+16)
  u16* myP = shP + wave * (16 * 64);
  unsigned char* myM = shM + wave * 2048;

  // Mask DMA lane mapping: lane -> (row, 32B half of a 64B mask row)
  const int mrw = lane >> 1, mhalf = lane & 1;
  const unsigned char* mrow0 =
      mask + (size_t)b * S_LEN * S_LEN + (size_t)(s0 + mrw) * S_LEN + mhalf * 32;

  // Q A-fragments, HD=64 -> two 16x32 k-steps (loaded once)
  Frag aq[2];
  {
    const u16* p = Qh + (size_t)(b * S_LEN + s0 + l15) * D_DIM + h * HD_;
#pragma unroll
    for (int kk = 0; kk < 2; ++kk) {
      aq[kk].q[0] = *(const uint4*)(p + kk * 32 + 8 * g);
      aq[kk].q[1] = *(const uint4*)(p + kk * 32 + 16 + 8 * g);
    }
  }

  float mrowv[8], lrow[8];
  v8f O[4];
#pragma unroll
  for (int r = 0; r < 8; ++r) { mrowv[r] = -1e30f; lrow[r] = 0.f; }
#pragma unroll
  for (int f = 0; f < 4; ++f)
#pragma unroll
    for (int r = 0; r < 8; ++r) O[f][r] = 0.f;

  // Kick off async mask DMA for key block 0 into buffer 0.
  {
    unsigned loff = (unsigned)(uintptr_t)(myM + mrw * 64 + mhalf * 32);
    unsigned long long ga = (unsigned long long)(uintptr_t)mrow0;
    async_b128(loff, ga);
    async_b128(loff + 16, ga + 16);
  }

  for (int kb = 0; kb < S_LEN; kb += 64) {
    const int buf = (kb >> 6) & 1;
    const bool more = (kb + 64) < S_LEN;
    if (more) {  // issue next block's mask DMA before this block's compute
      unsigned loff = (unsigned)(uintptr_t)(myM + (buf ^ 1) * 1024 + mrw * 64 + mhalf * 32);
      unsigned long long ga = (unsigned long long)(uintptr_t)(mrow0 + kb + 64);
      async_b128(loff, ga);
      async_b128(loff + 16, ga + 16);
      __builtin_prefetch(Kh + (size_t)(b * S_LEN + kb + 64 + l15) * D_DIM + h * HD_, 0, 1);
      __builtin_prefetch(Vt + ((size_t)(b * D_DIM + h * HD_ + l15)) * S_LEN + kb + 64, 0, 1);
    }

    // ---- S = Q * K^T  (K row-major == Kt column-major for B-frags)
    v8f Sf[4];
#pragma unroll
    for (int nf = 0; nf < 4; ++nf) {
      const u16* kp = Kh + (size_t)(b * S_LEN + kb + nf * 16 + l15) * D_DIM + h * HD_;
      v8f c;
#pragma unroll
      for (int r = 0; r < 8; ++r) c[r] = 0.f;
#pragma unroll
      for (int kk = 0; kk < 2; ++kk) {
        Frag bk_;
        bk_.q[0] = *(const uint4*)(kp + kk * 32 + 8 * g);
        bk_.q[1] = *(const uint4*)(kp + kk * 32 + 16 + 8 * g);
        c = wmma_f16(aq[kk], bk_, c);
      }
      Sf[nf] = c;
    }

    // current block's mask DMA must have landed (in-order completion:
    // allow the 2 just-issued copies for the next block to stay in flight)
    if (more) asm volatile("s_wait_asynccnt 0x2" ::: "memory");
    else      asm volatile("s_wait_asynccnt 0x0" ::: "memory");

    // ---- scale + mask-to-zero (from LDS), running row max
    const unsigned char* mb = myM + buf * 1024;
    float bm[8];
#pragma unroll
    for (int r = 0; r < 8; ++r) bm[r] = mrowv[r];
#pragma unroll
    for (int nf = 0; nf < 4; ++nf) {
      const int kc = nf * 16 + l15;
#pragma unroll
      for (int r = 0; r < 8; ++r) {
        float sc = mb[(8 * g + r) * 64 + kc] ? 0.0f : Sf[nf][r] * 0.125f;
        Sf[nf][r] = sc;
        bm[r] = fmaxf(bm[r], sc);
      }
    }
#pragma unroll
    for (int r = 0; r < 8; ++r)
#pragma unroll
      for (int off = 1; off < 16; off <<= 1)
        bm[r] = fmaxf(bm[r], __shfl_xor(bm[r], off, 32));  // within 16-lane half
    float corr[8];
#pragma unroll
    for (int r = 0; r < 8; ++r) {
      corr[r] = __expf(mrowv[r] - bm[r]);
      mrowv[r] = bm[r];
      lrow[r] *= corr[r];
    }
#pragma unroll
    for (int f = 0; f < 4; ++f)
#pragma unroll
      for (int r = 0; r < 8; ++r) O[f][r] *= corr[r];

    // ---- P = exp(S - m), row sums, stage P (f16) to LDS in A-layout order
    float rs[8];
#pragma unroll
    for (int r = 0; r < 8; ++r) rs[r] = 0.f;
#pragma unroll
    for (int nf = 0; nf < 4; ++nf)
#pragma unroll
      for (int r = 0; r < 8; ++r) {
        float p = __expf(Sf[nf][r] - mrowv[r]);
        rs[r] += p;
        myP[(8 * g + r) * 64 + nf * 16 + l15] = f2h(p);
      }
#pragma unroll
    for (int r = 0; r < 8; ++r) {
#pragma unroll
      for (int off = 1; off < 16; off <<= 1) rs[r] += __shfl_xor(rs[r], off, 32);
      lrow[r] += rs[r];
    }
    // retire all our DS ops (P stores + mask byte reads) before reading P /
    // before next iteration's async DMA can overwrite the old mask buffer
    asm volatile("s_wait_dscnt 0x0" ::: "memory");

    // ---- O += P * V   (Vt is [B, D, S] so each B-frag is two 16B runs along S)
#pragma unroll
    for (int kk = 0; kk < 2; ++kk) {
      Frag ap;
      const u16* pp = myP + l15 * 64 + kk * 32 + 8 * g;
      ap.q[0] = *(const uint4*)pp;
      ap.q[1] = *(const uint4*)(pp + 16);
#pragma unroll
      for (int f = 0; f < 4; ++f) {
        const u16* vp = Vt + ((size_t)(b * D_DIM + h * HD_ + f * 16 + l15)) * S_LEN
                        + kb + kk * 32 + 8 * g;
        Frag bv_;
        bv_.q[0] = *(const uint4*)vp;
        bv_.q[1] = *(const uint4*)(vp + 16);
        O[f] = wmma_f16(ap, bv_, O[f]);
      }
    }
  }

  // ---- epilogue: O / l, write fp32 [B, S, H*HD]
#pragma unroll
  for (int f = 0; f < 4; ++f) {
    const int d = h * HD_ + f * 16 + l15;
#pragma unroll
    for (int r = 0; r < 8; ++r) {
      const int qg = s0 + 8 * g + r;
      out[(size_t)(b * S_LEN + qg) * D_DIM + d] = O[f][r] / lrow[r];
    }
  }
}

// ---------------------------------------------------------------- launcher
extern "C" void kernel_launch(void* const* d_in, const int* in_sizes, int n_in,
                              void* d_out, int out_size, void* d_ws, size_t ws_size,
                              hipStream_t stream) {
  (void)in_sizes; (void)n_in; (void)out_size; (void)ws_size;
  const float* x          = (const float*)d_in[0];
  const unsigned char* mk = (const unsigned char*)d_in[1];  // jnp bool = 1 byte
  const float* Wq = (const float*)d_in[2];
  const float* bq = (const float*)d_in[3];
  const float* Wk = (const float*)d_in[4];
  const float* bk = (const float*)d_in[5];
  const float* Wv = (const float*)d_in[6];
  const float* bv = (const float*)d_in[7];
  float* out = (float*)d_out;

  char* ws = (char*)d_ws;
  u16* xh = (u16*)ws;                      // x f16             : 8 MB
  u16* Wt = (u16*)(ws + (8u  << 20));      // W^T f16 x3        : 6 MB
  u16* Qh = (u16*)(ws + (14u << 20));      // Q f16 [B*S, D]    : 8 MB
  u16* Kh = (u16*)(ws + (22u << 20));      // K f16 [B*S, D]    : 8 MB
  u16* Vt = (u16*)(ws + (30u << 20));      // V f16 [B, D, S]   : 8 MB

  const int nx = BB * S_LEN * D_DIM;                       // 4.19M
  cvt_f32_f16<<<nx / (256 * 4), 256, 0, stream>>>(x, xh, nx);
  wtrans<<<dim3(D_DIM / 32, D_DIM / 32, 3), dim3(32, 8), 0, stream>>>(Wq, Wk, Wv, Wt);
  qkv_gemm<<<dim3(BB * S_LEN / 128, D_DIM / 64, 3), 256, 0, stream>>>(
      xh, Wt, bq, bk, bv, Qh, Kh, Vt);
  attn<<<dim3(S_LEN / 64, BB * NH), 128, 0, stream>>>(Qh, Kh, Vt, mk, out);
}
